// MultiScaleRetentionCompat_37400575213787
// MI455X (gfx1250) — matrix operand order repaired and checked by reading, not compiled, not run on hardware
//
#include <hip/hip_runtime.h>
#include <math.h>

// ---------------------------------------------------------------------------
// Problem constants (from reference): B=4, S=4096, D_MODEL=1024, H=16, HD=64
// ---------------------------------------------------------------------------
static constexpr int Mrows = 4 * 4096;   // B*S = 16384
static constexpr int Ncols = 1024;       // D_MODEL
static constexpr int Kdim  = 1024;       // D_MODEL
static constexpr int NCH   = 16;         // scan chunks
static constexpr int CHL   = 4096 / NCH; // 256 steps per chunk
static constexpr float LN_EPS = 1e-5f;

typedef __bf16 bf16_t;
typedef __attribute__((ext_vector_type(8)))  bf16_t v8bf;
typedef __attribute__((ext_vector_type(16))) bf16_t v16bf;
typedef __attribute__((ext_vector_type(8)))  float  v8f;
typedef __attribute__((ext_vector_type(4)))  unsigned int u32x4;
typedef __attribute__((ext_vector_type(8)))  unsigned int u32x8;

union FragBF {
    v16bf v;
    v8bf  h[2];
};

// Padded LDS row: 32 bf16 (64B) + TDM pad of 4 DWORDs (16B) -> 40 elements.
// Lane r reads 16B at byte r*80 -> bank r*20 mod 64, distinct for r=0..15.
static constexpr int ROWE = 40;

// ---------------------------------------------------------------------------
// f32 -> bf16 convert (grid-stride)
// ---------------------------------------------------------------------------
__global__ void cvt_f32_bf16(const float* __restrict__ in,
                             bf16_t* __restrict__ out, int n) {
    for (int i = blockIdx.x * blockDim.x + threadIdx.x; i < n;
         i += gridDim.x * blockDim.x) {
        out[i] = (bf16_t)in[i];
    }
}

// ---------------------------------------------------------------------------
// Issue one TDM copy: 128 x 32 bf16 tile (row-major, row stride = K elements)
// from global to LDS with 16B padding per 64B row.
// D# layout per CDNA5 ISA 8.3/8.4 (2-D tensor -> groups 2/3 NULL).
// ---------------------------------------------------------------------------
__device__ __forceinline__ void tdm_load_tile(unsigned lds_off,
                                              unsigned long long gaddr,
                                              unsigned tensorDim1, unsigned K) {
    u32x4 g0;
    u32x8 g1;
    g0[0] = 1u;                                   // count=1, user desc
    g0[1] = lds_off;                              // lds_addr (bytes)
    g0[2] = (unsigned)(gaddr & 0xFFFFFFFFu);      // global_addr[31:0]
    g0[3] = (unsigned)((gaddr >> 32) & 0x01FFFFFFu) // global_addr[56:32]
            | 0x80000000u;                        // type=2 ("image")
    g1[0] = (1u << 16)                            // data_size = 1 (2 bytes)
            | (1u << 20)                          // pad_enable
            | (3u << 22)                          // pad_interval: 16 DWORDs
            | (3u << 25);                         // pad_amount: 4 DWORDs
    g1[1] = (K & 0xFFFFu) << 16;                  // tensor_dim0[15:0]
    g1[2] = ((K >> 16) & 0xFFFFu)                 // tensor_dim0[31:16]
            | ((tensorDim1 & 0xFFFFu) << 16);     // tensor_dim1[15:0]
    g1[3] = ((tensorDim1 >> 16) & 0xFFFFu)        // tensor_dim1[31:16]
            | (32u << 16);                        // tile_dim0 = 32
    g1[4] = 128u;                                 // tile_dim1 = 128 (tile_dim2=0)
    g1[5] = K;                                    // tensor_dim0_stride[31:0]
    g1[6] = 0u;                                   // stride[47:32], dim1_stride lo
    g1[7] = 0u;
    asm volatile("tensor_load_to_lds %0, %1" :: "s"(g0), "s"(g1) : "memory");
}

// ---------------------------------------------------------------------------
// D = A[M,K] x W[N,K]^T via v_wmma_f32_16x16x32_bf16, TDM-staged through LDS.
// Block = 256 threads = 8 waves; block tile 128x128; wave tile 32(M) x 64(N).
// Double-buffered: wave 0 issues tensor_load_to_lds for tile k+1 while all
// waves compute tile k from LDS; s_wait_tensorcnt + barrier hand-off.
// EPI==0: store f32.  EPI==1: store bf16(SiLU(d)).
// ---------------------------------------------------------------------------
template <int EPI>
__global__ __launch_bounds__(256)
void wmma_gemm_xwT(const bf16_t* __restrict__ A,
                   const bf16_t* __restrict__ W,
                   void* __restrict__ outp,
                   int M, int N, int K) {
    __shared__ __align__(16) bf16_t ldsA[2][128 * ROWE];
    __shared__ __align__(16) bf16_t ldsB[2][128 * ROWE];

    const int lane = threadIdx.x & 31;
    const int wave = threadIdx.x >> 5;
    const int wm = wave & 3;   // 0..3 -> M
    const int wn = wave >> 2;  // 0..1 -> N
    const int mBase = blockIdx.y * 128;
    const int nBase = blockIdx.x * 128;

    const int rowSel = lane & 15;
    const int kbA = (lane >> 4) * 8;    // A frag: interleaved K halves
    const int kbB = (lane >> 4) * 8;    // B frag: interleaved K halves

    const unsigned long long aAddr0 =
        (unsigned long long)(uintptr_t)(A + (size_t)mBase * K);
    const unsigned long long bAddr0 =
        (unsigned long long)(uintptr_t)(W + (size_t)nBase * K);

    v8f acc[2][4] = {};

    int buf = 0;
    if (wave == 0) {
        tdm_load_tile((unsigned)(uintptr_t)&ldsA[0][0], aAddr0, (unsigned)M,
                      (unsigned)K);
        tdm_load_tile((unsigned)(uintptr_t)&ldsB[0][0], bAddr0, (unsigned)N,
                      (unsigned)K);
        __builtin_amdgcn_s_wait_tensorcnt(0);
    }
    __syncthreads();

    for (int k0 = 0; k0 < K; k0 += 32) {
        const int nbuf = buf ^ 1;
        const bool more = (k0 + 32) < K;
        if (more && wave == 0) {
            const unsigned long long koff = (unsigned long long)(k0 + 32) * 2;
            tdm_load_tile((unsigned)(uintptr_t)&ldsA[nbuf][0], aAddr0 + koff,
                          (unsigned)M, (unsigned)K);
            tdm_load_tile((unsigned)(uintptr_t)&ldsB[nbuf][0], bAddr0 + koff,
                          (unsigned)N, (unsigned)K);
        }

        // ---- compute tile `buf` from LDS ----
        FragBF a[2], b[4];
#pragma unroll
        for (int mi = 0; mi < 2; ++mi) {
            const bf16_t* p =
                &ldsA[buf][(wm * 32 + mi * 16 + rowSel) * ROWE + kbA];
            a[mi].h[0] = *(const v8bf*)(p);
            a[mi].h[1] = *(const v8bf*)(p + 16);
        }
#pragma unroll
        for (int ni = 0; ni < 4; ++ni) {
            const bf16_t* p =
                &ldsB[buf][(wn * 64 + ni * 16 + rowSel) * ROWE + kbB];
            b[ni].h[0] = *(const v8bf*)(p);
            b[ni].h[1] = *(const v8bf*)(p + 16);
        }
#pragma unroll
        for (int ni = 0; ni < 4; ++ni) {
            acc[0][ni] = __builtin_amdgcn_wmma_f32_16x16x32_bf16(
                false, a[0].v, false, b[ni].v, (short)0, acc[0][ni], false, false);
            acc[1][ni] = __builtin_amdgcn_wmma_f32_16x16x32_bf16(
                false, a[1].v, false, b[ni].v, (short)0, acc[1][ni], false, false);
        }

        if (more && wave == 0)
            __builtin_amdgcn_s_wait_tensorcnt(0);
        __syncthreads();
        buf = nbuf;
    }

    // ---- epilogue ----
    const int colD = lane & 15;
    const int rowD = (lane >> 4) * 8;
#pragma unroll
    for (int mi = 0; mi < 2; ++mi) {
#pragma unroll
        for (int ni = 0; ni < 4; ++ni) {
#pragma unroll
            for (int r = 0; r < 8; ++r) {
                const int row = mBase + wm * 32 + mi * 16 + rowD + r;
                const int col = nBase + wn * 64 + ni * 16 + colD;
                const float d = acc[mi][ni][r];
                if (EPI == 0) {
                    ((float*)outp)[(size_t)row * N + col] = d;
                } else {
                    const float s = d / (1.0f + __expf(-d));  // SiLU
                    ((bf16_t*)outp)[(size_t)row * N + col] = (bf16_t)s;
                }
            }
        }
    }
}

// ---------------------------------------------------------------------------
// Chunked EMA scan: s_t = lam*s_{t-1} + v_t over S, per (b, head, dim).
// Layout of carry/prefix: [b][chunk][h][d] == linear tid.
// ---------------------------------------------------------------------------
__global__ __launch_bounds__(256)
void scan_pass1(const float* __restrict__ v, const float* __restrict__ beta,
                float* __restrict__ carry) {
    const int tid = blockIdx.x * blockDim.x + threadIdx.x;  // 65536
    const int d = tid & 63;
    const int h = (tid >> 6) & 15;
    const int c = (tid >> 10) & 15;
    const int b = tid >> 14;
    const float lam = 1.0f / (1.0f + __expf(-beta[h]));
    const size_t base = ((size_t)b * 4096 + (size_t)c * CHL) * 1024 + h * 64 + d;
    float s = 0.0f;
    for (int i = 0; i < CHL; ++i)
        s = fmaf(lam, s, v[base + (size_t)i * 1024]);
    carry[tid] = s;
}

__global__ __launch_bounds__(256)
void scan_pass2(const float* __restrict__ carry, const float* __restrict__ beta,
                float* __restrict__ prefix) {
    const int tid = blockIdx.x * blockDim.x + threadIdx.x;  // 4096
    const int d = tid & 63;
    const int h = (tid >> 6) & 15;
    const int b = tid >> 10;
    const float lam  = 1.0f / (1.0f + __expf(-beta[h]));
    const float lamL = __powf(lam, (float)CHL);
    float run = 0.0f;
    for (int c = 0; c < NCH; ++c) {
        const int idx = ((b * NCH + c) * 16 + h) * 64 + d;
        prefix[idx] = run;
        run = fmaf(lamL, run, carry[idx]);
    }
}

__global__ __launch_bounds__(256)
void scan_pass3(const float* __restrict__ v, const float* __restrict__ q,
                const float* __restrict__ beta, const float* __restrict__ prefix,
                float* __restrict__ y /* may alias q */) {
    const int tid = blockIdx.x * blockDim.x + threadIdx.x;  // 65536
    const int d = tid & 63;
    const int h = (tid >> 6) & 15;
    const int c = (tid >> 10) & 15;
    const int b = tid >> 14;
    const float lam = 1.0f / (1.0f + __expf(-beta[h]));
    const size_t base = ((size_t)b * 4096 + (size_t)c * CHL) * 1024 + h * 64 + d;
    float s = prefix[tid];
    for (int i = 0; i < CHL; ++i) {
        const size_t idx = base + (size_t)i * 1024;
        s = fmaf(lam, s, v[idx]);
        y[idx] = q[idx] * s;
    }
}

// ---------------------------------------------------------------------------
// LayerNorm(row of 1024) * gamma + beta, then * gate -> bf16
// One 256-thread block (8 wave32) per row; shuffle + LDS reduction.
// ---------------------------------------------------------------------------
__global__ __launch_bounds__(256)
void ln_gate(const float* __restrict__ y, const bf16_t* __restrict__ gate,
             const float* __restrict__ gamma, const float* __restrict__ betaLn,
             bf16_t* __restrict__ out) {
    __shared__ float red[16];
    const int row = blockIdx.x;
    const float* yr = y + (size_t)row * 1024;
    float vreg[4];
    float sum = 0.0f, sq = 0.0f;
#pragma unroll
    for (int j = 0; j < 4; ++j) {
        vreg[j] = yr[threadIdx.x + j * 256];
        sum += vreg[j];
        sq  += vreg[j] * vreg[j];
    }
#pragma unroll
    for (int o = 16; o > 0; o >>= 1) {
        sum += __shfl_xor(sum, o, 32);
        sq  += __shfl_xor(sq, o, 32);
    }
    const int wave = threadIdx.x >> 5;
    const int lane = threadIdx.x & 31;
    if (lane == 0) { red[wave] = sum; red[8 + wave] = sq; }
    __syncthreads();
    if (threadIdx.x == 0) {
        float a = 0.0f, b2 = 0.0f;
        for (int w = 0; w < 8; ++w) { a += red[w]; b2 += red[8 + w]; }
        red[0] = a; red[8] = b2;
    }
    __syncthreads();
    const float mu   = red[0] * (1.0f / 1024.0f);
    const float var  = red[8] * (1.0f / 1024.0f) - mu * mu;
    const float rstd = rsqrtf(var + LN_EPS);
#pragma unroll
    for (int j = 0; j < 4; ++j) {
        const int col = threadIdx.x + j * 256;
        const size_t idx = (size_t)row * 1024 + col;
        const float g = (float)gate[idx];
        const float o = (vreg[j] - mu) * rstd * gamma[col] + betaLn[col];
        out[idx] = (bf16_t)(o * g);
    }
}

// ---------------------------------------------------------------------------
// Launch
// ---------------------------------------------------------------------------
extern "C" void kernel_launch(void* const* d_in, const int* in_sizes, int n_in,
                              void* d_out, int out_size, void* d_ws, size_t ws_size,
                              hipStream_t stream) {
    const float* x     = (const float*)d_in[0];
    const float* Wq    = (const float*)d_in[1];
    const float* Wv    = (const float*)d_in[2];
    const float* Wo    = (const float*)d_in[3];
    const float* Wg    = (const float*)d_in[4];
    const float* beta  = (const float*)d_in[5];
    const float* gamma = (const float*)d_in[6];
    const float* bln   = (const float*)d_in[7];
    float* out = (float*)d_out;

    // Workspace carve-up
    char* w = (char*)d_ws;
    size_t cur = 0;
    auto alloc = [&](size_t bytes) {
        void* p = w + cur;
        cur += (bytes + 255) & ~(size_t)255;
        return p;
    };
    const size_t nX = (size_t)Mrows * Kdim;      // 16.7M elems
    const size_t nW = (size_t)Ncols * Kdim;      // 1.05M elems
    bf16_t* xb   = (bf16_t*)alloc(nX * 2);
    bf16_t* wqb  = (bf16_t*)alloc(nW * 2);
    bf16_t* wvb  = (bf16_t*)alloc(nW * 2);
    bf16_t* wgb  = (bf16_t*)alloc(nW * 2);
    bf16_t* wob  = (bf16_t*)alloc(nW * 2);
    float*  qbuf = (float*)alloc(nX * 4);
    float*  vbuf = (float*)alloc(nX * 4);
    bf16_t* gbuf = (bf16_t*)alloc(nX * 2);
    bf16_t* hb   = (bf16_t*)alloc(nX * 2);
    float*  carry  = (float*)alloc((size_t)4 * NCH * 16 * 64 * 4);
    float*  prefix = (float*)alloc((size_t)4 * NCH * 16 * 64 * 4);
    (void)ws_size; (void)in_sizes; (void)n_in; (void)out_size;

    // 1. converts
    cvt_f32_bf16<<<8192, 256, 0, stream>>>(x, xb, (int)nX);
    cvt_f32_bf16<<<1024, 256, 0, stream>>>(Wq, wqb, (int)nW);
    cvt_f32_bf16<<<1024, 256, 0, stream>>>(Wv, wvb, (int)nW);
    cvt_f32_bf16<<<1024, 256, 0, stream>>>(Wg, wgb, (int)nW);
    cvt_f32_bf16<<<1024, 256, 0, stream>>>(Wo, wob, (int)nW);

    // 2. projection GEMMs (q, v, gate=SiLU)
    dim3 gGrid(Ncols / 128, Mrows / 128);  // (8, 128)
    wmma_gemm_xwT<0><<<gGrid, 256, 0, stream>>>(xb, wqb, qbuf, Mrows, Ncols, Kdim);
    wmma_gemm_xwT<0><<<gGrid, 256, 0, stream>>>(xb, wvb, vbuf, Mrows, Ncols, Kdim);
    wmma_gemm_xwT<1><<<gGrid, 256, 0, stream>>>(xb, wgb, gbuf, Mrows, Ncols, Kdim);

    // 3. chunked EMA scan + q*state (in-place over qbuf)
    scan_pass1<<<256, 256, 0, stream>>>(vbuf, beta, carry);
    scan_pass2<<<16, 256, 0, stream>>>(carry, beta, prefix);
    scan_pass3<<<256, 256, 0, stream>>>(vbuf, qbuf, beta, prefix, qbuf);

    // 4. LayerNorm + gate -> bf16
    ln_gate<<<Mrows, 256, 0, stream>>>(qbuf, gbuf, gamma, bln, hb);

    // 5. output projection
    wmma_gemm_xwT<0><<<gGrid, 256, 0, stream>>>(hb, wob, out, Mrows, Ncols, Kdim);
}
